// DNN_82119774700170
// MI455X (gfx1250) — matrix operand order repaired and checked by reading, not compiled, not run on hardware
//
#include <hip/hip_runtime.h>
#include <hip/hip_bf16.h>
#include <math.h>

typedef __attribute__((ext_vector_type(16))) _Float16 v16h;
typedef __attribute__((ext_vector_type(8)))  _Float16 v8h;
typedef __attribute__((ext_vector_type(4)))  _Float16 v4h;
typedef __attribute__((ext_vector_type(8)))  float    v8f;

#define N_AGENTS 32
#define OBS_DIM  128
#define HIDDEN   128
#define N_LAYERS 2
#define ACT_DIM  16
#define BATCH    16384
#define BT       32        // batch rows per block (2 x 16-row subtiles)

// ---- workspace layout (bytes) ----
#define WOBS_OFF  0
#define WOBS_BYTES (N_AGENTS * HIDDEN * OBS_DIM * 2)            // 1,048,576
#define WHP_OFF   (WOBS_OFF + WOBS_BYTES)
#define WHP_BYTES (N_LAYERS * N_AGENTS * HIDDEN * HIDDEN * 2)   // 2,097,152
#define WCP_OFF   (WHP_OFF + WHP_BYTES)
#define WCP_BYTES WHP_BYTES
#define WQ_OFF    (WCP_OFF + WCP_BYTES)
#define WQ_BYTES  (N_AGENTS * ACT_DIM * HIDDEN * 2)             // 131,072
#define BHC_OFF   (WQ_OFF + WQ_BYTES)
#define BHC_BYTES (N_LAYERS * N_AGENTS * HIDDEN * 4)            // 32,768

static __device__ __forceinline__ v8f wmma_f16(v16h a, v16h b, v8f c) {
    return __builtin_amdgcn_wmma_f32_16x16x32_f16(
        /*neg_a=*/false, a, /*neg_b=*/false, b,
        /*c_mod=*/(short)0, c, /*reuse_a=*/false, /*reuse_b=*/false);
}

// A fragment (16x32 f16, MxK) per CDNA5 layout:
// lane: m = lane&15, khalf = lane>>4
// VGPR 0-3: K = khalf*8 + {0..7};  VGPR 4-7: K = khalf*8 + 16 + {0..7}
static __device__ __forceinline__ v16h load_a_frag(const _Float16* tile, int m0,
                                                   int k0, int lane) {
    int m    = m0 + (lane & 15);
    int base = k0 + ((lane >> 4) << 3);
    const _Float16* p = tile + m * HIDDEN + base;
    v8h lo = *(const v8h*)(p);
    v8h hi = *(const v8h*)(p + 16);
    v16h a;
#pragma unroll
    for (int i = 0; i < 8; ++i) { a[i] = lo[i]; a[i + 8] = hi[i]; }
    return a;
}

// B fragment (32x16 f16, KxN) for W stored row-major [n][k] (torch Linear [out,in]):
// lane: n = lane&15, khalf = lane>>4 -> K = khalf*8 + {0..7}, khalf*8 + 16 + {0..7} (same as A)
static __device__ __forceinline__ v16h load_b_frag(const _Float16* W, int n0,
                                                   int k0, int lane) {
    int n = n0 + (lane & 15);
    int k = k0 + ((lane >> 4) << 3);
    const _Float16* p = W + n * HIDDEN + k;
    v8h lo = *(const v8h*)(p);
    v8h hi = *(const v8h*)(p + 16);
    v16h b;
#pragma unroll
    for (int i = 0; i < 8; ++i) { b[i] = lo[i]; b[i + 8] = hi[i]; }
    return b;
}

// store one 16x16 f32 D tile into row-major f16 LDS tile
static __device__ __forceinline__ void store_d_tile(_Float16* tile, int m0, int n0,
                                                    v8f acc, int lane) {
#pragma unroll
    for (int r = 0; r < 8; ++r) {
        int row = m0 + r + ((lane >> 4) << 3);
        tile[row * HIDDEN + n0 + (lane & 15)] = (_Float16)acc[r];
    }
}

// ---- prep: fold weights to f16 (Wh' = W_H - W_C/31, Wc' = W_C/31), combine biases
__global__ void __launch_bounds__(256) prep_kernel(
    const float* __restrict__ W_obs, const float* __restrict__ W_H,
    const float* __restrict__ W_C,  const float* __restrict__ b_H,
    const float* __restrict__ b_C,  const float* __restrict__ W_q,
    char* __restrict__ ws) {
    _Float16* WobsH = (_Float16*)(ws + WOBS_OFF);
    _Float16* WhP   = (_Float16*)(ws + WHP_OFF);
    _Float16* WcP   = (_Float16*)(ws + WCP_OFF);
    _Float16* WqH   = (_Float16*)(ws + WQ_OFF);
    float*    bHC   = (float*)   (ws + BHC_OFF);

    const int N0 = N_AGENTS * HIDDEN * OBS_DIM;            // 524288
    const int N1 = N_LAYERS * N_AGENTS * HIDDEN * HIDDEN;  // 1048576
    const int N2 = N_AGENTS * ACT_DIM * HIDDEN;            // 65536
    const int N3 = N_LAYERS * N_AGENTS * HIDDEN;           // 8192
    const int NT = N0 + N1 + N2 + N3;
    const float inv31 = 1.0f / 31.0f;

    for (int i = blockIdx.x * blockDim.x + threadIdx.x; i < NT;
         i += gridDim.x * blockDim.x) {
        if (i < N0) {
            WobsH[i] = (_Float16)W_obs[i];
        } else if (i < N0 + N1) {
            int j = i - N0;
            float wc = W_C[j] * inv31;
            WhP[j] = (_Float16)(W_H[j] - wc);
            WcP[j] = (_Float16)wc;
        } else if (i < N0 + N1 + N2) {
            int j = i - N0 - N1;
            WqH[j] = (_Float16)W_q[j];
        } else {
            int j = i - N0 - N1 - N2;
            bHC[j] = b_H[j] + b_C[j];
        }
    }
}

// ---- main: one block = 32 batch rows x all 32 agents, 8 waves
// One wave owns whole agents (both 16-row subtiles) so every B fragment
// loaded from L2 feeds TWO wmmas -> halves weight traffic.
__global__ void __launch_bounds__(256) commnet_kernel(
    const float* __restrict__ obs, const float* __restrict__ b_obs,
    const float* __restrict__ b_q, const char* __restrict__ ws,
    float* __restrict__ out) {
    extern __shared__ _Float16 smem[];
    _Float16* hbuf = smem;                                // [32][BT][128]
    _Float16* sumb = smem + N_AGENTS * BT * HIDDEN;       // [BT][128]

    const _Float16* WobsH = (const _Float16*)(ws + WOBS_OFF);
    const _Float16* WhP   = (const _Float16*)(ws + WHP_OFF);
    const _Float16* WcP   = (const _Float16*)(ws + WCP_OFF);
    const _Float16* WqH   = (const _Float16*)(ws + WQ_OFF);
    const float*    bHC   = (const float*)   (ws + BHC_OFF);

    const int tid  = threadIdx.x;
    const int lane = tid & 31;
    const int wave = tid >> 5;
    const int r0   = blockIdx.x * BT;

    // ---- stage obs (f32 -> f16) into hbuf ----
    for (int c = tid; c < N_AGENTS * BT * (OBS_DIM / 4); c += 256) {
        int a   = c >> 10;
        int rem = c & 1023;
        int r   = rem >> 5;
        int cc  = (rem & 31) << 2;
        const float4 v =
            *(const float4*)&obs[((size_t)a * BATCH + r0 + r) * OBS_DIM + cc];
        v4h h;
        h[0] = (_Float16)v.x; h[1] = (_Float16)v.y;
        h[2] = (_Float16)v.z; h[3] = (_Float16)v.w;
        *(v4h*)&hbuf[(a * BT + r) * HIDDEN + cc] = h;
    }
    __syncthreads();

    // ---- obs projection: h = relu(obs @ W_obs^T + b_obs) ----
    for (int ag = wave; ag < N_AGENTS; ag += 8) {
        _Float16* htile = &hbuf[ag * BT * HIDDEN];
        v16h A0[4], A1[4];
#pragma unroll
        for (int kf = 0; kf < 4; ++kf) {
            A0[kf] = load_a_frag(htile, 0,  kf * 32, lane);
            A1[kf] = load_a_frag(htile, 16, kf * 32, lane);
        }
        const _Float16* W = WobsH + ag * HIDDEN * OBS_DIM;
#pragma unroll
        for (int j = 0; j < 8; ++j) {
            float bv = b_obs[ag * HIDDEN + (j << 4) + (lane & 15)];
            v8f acc0, acc1;
#pragma unroll
            for (int r = 0; r < 8; ++r) { acc0[r] = bv; acc1[r] = bv; }
#pragma unroll
            for (int kf = 0; kf < 4; ++kf) {
                v16h B = load_b_frag(W, j << 4, kf * 32, lane);
                acc0 = wmma_f16(A0[kf], B, acc0);
                acc1 = wmma_f16(A1[kf], B, acc1);
            }
#pragma unroll
            for (int r = 0; r < 8; ++r) {
                acc0[r] = acc0[r] > 0.0f ? acc0[r] : 0.0f;
                acc1[r] = acc1[r] > 0.0f ? acc1[r] : 0.0f;
            }
            store_d_tile(htile, 0,  j << 4, acc0, lane);
            store_d_tile(htile, 16, j << 4, acc1, lane);
        }
    }
    __syncthreads();

    // ---- comm layers: h = tanh(h @ Wh'^T + sum @ Wc'^T + bHC) ----
    for (int l = 0; l < N_LAYERS; ++l) {
        // build sum over agents (f32 accumulate, f16 store) with b128 LDS reads
        for (int idx = tid; idx < BT * (HIDDEN / 8); idx += 256) {
            int r  = idx >> 4;
            int c8 = (idx & 15) << 3;
            float s[8];
#pragma unroll
            for (int i = 0; i < 8; ++i) s[i] = 0.0f;
#pragma unroll
            for (int a = 0; a < N_AGENTS; ++a) {
                v8h v = *(const v8h*)&hbuf[(a * BT + r) * HIDDEN + c8];
#pragma unroll
                for (int i = 0; i < 8; ++i) s[i] += (float)v[i];
            }
            v8h o;
#pragma unroll
            for (int i = 0; i < 8; ++i) o[i] = (_Float16)s[i];
            *(v8h*)&sumb[r * HIDDEN + c8] = o;
        }
        __syncthreads();

        for (int ag = wave; ag < N_AGENTS; ag += 8) {
            _Float16* htile = &hbuf[ag * BT * HIDDEN];
            v16h Ah0[4], Ah1[4], As0[4], As1[4];
#pragma unroll
            for (int kf = 0; kf < 4; ++kf) {
                Ah0[kf] = load_a_frag(htile, 0,  kf * 32, lane);
                Ah1[kf] = load_a_frag(htile, 16, kf * 32, lane);
                As0[kf] = load_a_frag(sumb,  0,  kf * 32, lane);
                As1[kf] = load_a_frag(sumb,  16, kf * 32, lane);
            }
            const _Float16* Wh = WhP + (size_t)(l * N_AGENTS + ag) * HIDDEN * HIDDEN;
            const _Float16* Wc = WcP + (size_t)(l * N_AGENTS + ag) * HIDDEN * HIDDEN;
            const float* bias  = bHC + (l * N_AGENTS + ag) * HIDDEN;
#pragma unroll
            for (int j = 0; j < 8; ++j) {
                float bv = bias[(j << 4) + (lane & 15)];
                v8f acc0, acc1;
#pragma unroll
                for (int r = 0; r < 8; ++r) { acc0[r] = bv; acc1[r] = bv; }
#pragma unroll
                for (int kf = 0; kf < 4; ++kf) {
                    v16h B = load_b_frag(Wh, j << 4, kf * 32, lane);
                    acc0 = wmma_f16(Ah0[kf], B, acc0);
                    acc1 = wmma_f16(Ah1[kf], B, acc1);
                }
#pragma unroll
                for (int kf = 0; kf < 4; ++kf) {
                    v16h B = load_b_frag(Wc, j << 4, kf * 32, lane);
                    acc0 = wmma_f16(As0[kf], B, acc0);
                    acc1 = wmma_f16(As1[kf], B, acc1);
                }
#pragma unroll
                for (int r = 0; r < 8; ++r) {
                    acc0[r] = tanhf(acc0[r]);
                    acc1[r] = tanhf(acc1[r]);
                }
                store_d_tile(htile, 0,  j << 4, acc0, lane);
                store_d_tile(htile, 16, j << 4, acc1, lane);
            }
        }
        __syncthreads();
    }

    // ---- q head: out = h @ W_q^T + b_q  (N = 16 -> one n-tile) ----
    for (int ag = wave; ag < N_AGENTS; ag += 8) {
        const _Float16* htile = &hbuf[ag * BT * HIDDEN];
        v16h A0[4], A1[4];
#pragma unroll
        for (int kf = 0; kf < 4; ++kf) {
            A0[kf] = load_a_frag(htile, 0,  kf * 32, lane);
            A1[kf] = load_a_frag(htile, 16, kf * 32, lane);
        }
        const _Float16* Wq = WqH + ag * ACT_DIM * HIDDEN;
        float bv = b_q[ag * ACT_DIM + (lane & 15)];
        v8f acc0, acc1;
#pragma unroll
        for (int r = 0; r < 8; ++r) { acc0[r] = bv; acc1[r] = bv; }
#pragma unroll
        for (int kf = 0; kf < 4; ++kf) {
            v16h B = load_b_frag(Wq, 0, kf * 32, lane);
            acc0 = wmma_f16(A0[kf], B, acc0);
            acc1 = wmma_f16(A1[kf], B, acc1);
        }
#pragma unroll
        for (int r = 0; r < 8; ++r) {
            int rowm = r + ((lane >> 4) << 3);
            out[((size_t)ag * BATCH + r0 + rowm) * ACT_DIM + (lane & 15)]      = acc0[r];
            out[((size_t)ag * BATCH + r0 + 16 + rowm) * ACT_DIM + (lane & 15)] = acc1[r];
        }
    }
}

extern "C" void kernel_launch(void* const* d_in, const int* in_sizes, int n_in,
                              void* d_out, int out_size, void* d_ws, size_t ws_size,
                              hipStream_t stream) {
    const float* obs   = (const float*)d_in[0];
    const float* W_obs = (const float*)d_in[1];
    const float* b_obs = (const float*)d_in[2];
    const float* W_H   = (const float*)d_in[3];
    const float* b_H   = (const float*)d_in[4];
    const float* W_C   = (const float*)d_in[5];
    const float* b_C   = (const float*)d_in[6];
    const float* W_q   = (const float*)d_in[7];
    const float* b_q   = (const float*)d_in[8];
    float* out = (float*)d_out;
    char*  ws  = (char*)d_ws;

    // fold/convert weights to f16 in workspace
    const int NT = N_AGENTS * HIDDEN * OBS_DIM + N_LAYERS * N_AGENTS * HIDDEN * HIDDEN +
                   N_AGENTS * ACT_DIM * HIDDEN + N_LAYERS * N_AGENTS * HIDDEN;
    prep_kernel<<<(NT + 255) / 256, 256, 0, stream>>>(W_obs, W_H, W_C, b_H, b_C, W_q, ws);

    // main fused kernel: 512 blocks x 256 threads, 264KB dynamic LDS
    const size_t smem_bytes =
        (size_t)(N_AGENTS * BT * HIDDEN + BT * HIDDEN) * sizeof(_Float16);
    commnet_kernel<<<BATCH / BT, 256, smem_bytes, stream>>>(obs, b_obs, b_q, ws, out);
}
